// Painter_Varient_33509334843902
// MI455X (gfx1250) — matrix operand order repaired and checked by reading, not compiled, not run on hardware
//
#include <hip/hip_runtime.h>
#include <hip/hip_bf16.h>

// ---------------------------------------------------------------------------
// MI455X (gfx1250) fused windowed-attention block.
//   QKV GEMM (bf16 WMMA, fp32 acc, 128x128 tiles) -> fused per-(head,batch)
//   attention in 320KB-class LDS (scores never touch HBM) -> proj GEMM.
// Workspace: ~111 MB bf16 staging (x buffer reused for attention output).
// ---------------------------------------------------------------------------

typedef __bf16 bf16;
typedef __attribute__((ext_vector_type(16))) __bf16 v16bf;
typedef __attribute__((ext_vector_type(8)))  __bf16 v8bf;
typedef __attribute__((ext_vector_type(8)))  float  v8f;

#define NHD  16      // heads
#define BSZ  64      // batch
#define HH   14
#define WW   14
#define NT   196     // tokens = 14*14
#define DIM  1024
#define HD   64      // head dim
#define PADM 208     // 13*16  (padded token count for 16x16 tiles)
#define SNC  224     // 7*32   (padded score row length for bf16 WMMA K-steps)

union bfpair { uint32_t u; bf16 h[2]; };

static __device__ __forceinline__ v16bf make_frag(const bf16* lo, const bf16* hi) {
  v8bf a = *(const v8bf*)lo;
  v8bf b = *(const v8bf*)hi;
  v16bf r;
#pragma unroll
  for (int i = 0; i < 8; ++i) { r[i] = a[i]; r[8 + i] = b[i]; }
  return r;
}

static __device__ __forceinline__ v8f wmma_bf16(v16bf a, v16bf b, v8f c) {
  // v_wmma_f32_16x16x32_bf16  (neg_a, A, neg_b, B, c_mod, C, reuse_a, reuse_b)
  return __builtin_amdgcn_wmma_f32_16x16x32_bf16(false, a, false, b, (short)0, c,
                                                 false, false);
}

// ---------------------------------------------------------------------------
// fp32 -> bf16 pack, 2 elements / thread (all sizes here are even)
// ---------------------------------------------------------------------------
__global__ __launch_bounds__(256) void cvt_f32_bf16(const float* __restrict__ src,
                                                    bf16* __restrict__ dst, int n2) {
  int i = blockIdx.x * 256 + threadIdx.x;
  if (i < n2) {
    float2 f = ((const float2*)src)[i];
    bfpair cv;
    cv.h[0] = (bf16)f.x;
    cv.h[1] = (bf16)f.y;
    ((uint32_t*)dst)[i] = cv.u;
  }
}

// ---------------------------------------------------------------------------
// Tiled bf16 WMMA GEMM:  C(MxN) = A(MxK) * B(KxN) + bias
//   Workgroup tile 128x128, BLOCK_K=32, 8 waves; wave w owns rows [16w,16w+16)
//   x all 128 cols (8 accumulators, 8 WMMAs per A-fragment). A tile and B tile
//   (transposed) staged in LDS so every fragment load is 2x ds_read_b128.
//   Next K-tile is prefetched (global_prefetch_b8) while current is consumed.
// MODE 0: QKV epilogue — scatter bf16 into q/k/v [head*BSZ+b][t][c]
// MODE 1: proj epilogue — fp32 store to out
// ---------------------------------------------------------------------------
template <int MODE>
__global__ __launch_bounds__(256) void gemm_bf16(
    const bf16* __restrict__ A, const bf16* __restrict__ Bw,
    const float* __restrict__ bias, float* __restrict__ outF,
    bf16* __restrict__ qb, bf16* __restrict__ kb, bf16* __restrict__ vb,
    int M, int N, int K) {
  __shared__ bf16 aT[128 * 32];   // [row][k]   8 KB
  __shared__ bf16 bT[128 * 32];   // [n][k]     8 KB  (transposed)

  const int tid   = threadIdx.x;
  const int wave  = tid >> 5;
  const int lane  = tid & 31;
  const int lhalf = lane >> 4;
  const int lmod  = lane & 15;
  const int m0 = blockIdx.x * 128;
  const int n0 = blockIdx.y * 128;

  v8f acc[8];
#pragma unroll
  for (int i = 0; i < 8; ++i) acc[i] = (v8f){0, 0, 0, 0, 0, 0, 0, 0};

  for (int kk = 0; kk < K; kk += 32) {
    // hint-prefetch next K-tile while this one is loaded/consumed
    if (kk + 32 < K) {
      __builtin_prefetch(A + (size_t)(m0 + (tid & 127)) * K + kk + 32, 0, 3);
      __builtin_prefetch(Bw + (size_t)(kk + 32 + (tid & 31)) * N + n0 + (tid >> 5) * 16,
                         0, 3);
    }
    __syncthreads();
    // A tile 128x32 (dword = 2 bf16, coalesced rows): 8 dwords / thread
    {
      const uint32_t* Ag = (const uint32_t*)(A + (size_t)m0 * K + kk);
      const int rs = K >> 1;
      for (int i = tid; i < 128 * 16; i += 256) {
        int row = i >> 4, cp = i & 15;
        ((uint32_t*)aT)[row * 16 + cp] = Ag[row * rs + cp];
      }
    }
    // B tile 32x128 loaded coalesced, stored transposed [n][k]
    for (int i = tid; i < 32 * 64; i += 256) {
      int k = i >> 6, np = i & 63;
      bfpair cv;
      cv.u = *(const uint32_t*)(Bw + (size_t)(kk + k) * N + n0 + np * 2);
      bT[(np * 2) * 32 + k]     = cv.h[0];
      bT[(np * 2 + 1) * 32 + k] = cv.h[1];
    }
    __syncthreads();

    const bf16* ar = aT + (wave * 16 + lmod) * 32 + lhalf * 8;
    v16bf af = make_frag(ar, ar + 16);
#pragma unroll
    for (int nt = 0; nt < 8; ++nt) {
      const bf16* br = bT + (nt * 16 + lmod) * 32 + lhalf * 8;
      acc[nt] = wmma_bf16(af, make_frag(br, br + 16), acc[nt]);
    }
  }

  // epilogue: D layout — vgpr v, lane l: row = v + 8*(l>>4), col = l&15
#pragma unroll
  for (int nt = 0; nt < 8; ++nt) {
#pragma unroll
    for (int v = 0; v < 8; ++v) {
      int r = m0 + wave * 16 + v + 8 * lhalf;
      int j = n0 + nt * 16 + lmod;
      float val = acc[nt][v] + bias[j];
      if (MODE == 1) {
        outF[(size_t)r * N + j] = val;
      } else {
        int which = j >> 10;           // 0=q 1=k 2=v
        int head  = (j >> 6) & 15;
        int c     = j & 63;
        int b     = r / NT;
        int t     = r - b * NT;
        bf16* dst = (which == 0) ? qb : (which == 1 ? kb : vb);
        dst[(((size_t)(head * BSZ + b)) * NT + t) * HD + c] = (bf16)val;
      }
    }
  }
}

// ---------------------------------------------------------------------------
// Fused attention, one workgroup per (head,batch) pair (1024 total).
// LDS (293,696 B dynamic — CDNA5 allows up to 320 KB/workgroup):
//   q_s [208][64] bf16 | k_s [208][64] bf16 | vt_s [64][208] bf16 (V^T)
//   S_s [208][224] f32 | relh/relw [208][16] f32 (zero-padded) | inv [208] f32
// The zero-padded q/k/vt rows and rel tables make every QK tile store exact
// zeros in padded rows (padded cols only meet zero V^T cols), so the QK epilogue is fully branch-free; the
// softmax 1/sum is folded into the PV epilogue (one fewer LDS pass).
// ---------------------------------------------------------------------------
__global__ __launch_bounds__(256) void attn_fused(
    const bf16* __restrict__ qg, const bf16* __restrict__ kg,
    const bf16* __restrict__ vg, const float* __restrict__ rel_h,
    const float* __restrict__ rel_w, bf16* __restrict__ out) {
  extern __shared__ char smem[];
  bf16*  q_s  = (bf16*)smem;                 // 26624 B
  bf16*  k_s  = q_s + PADM * HD;             // 26624 B
  bf16*  vt_s = k_s + PADM * HD;             // 26624 B
  float* S_s  = (float*)(vt_s + HD * PADM);  // 186368 B
  float* relh = S_s + PADM * SNC;            // 13312 B  [208][16]
  float* relw = relh + PADM * 16;            // 13312 B
  float* inv_s = relw + PADM * 16;           // 832 B

  const int tid   = threadIdx.x;
  const int hb    = blockIdx.x;   // head*BSZ + b
  const int head  = hb >> 6;
  const int b     = hb & 63;
  const int wave  = tid >> 5;
  const int lane  = tid & 31;
  const int lhalf = lane >> 4;
  const int lmod  = lane & 15;

  // zero scores + padding rows/cols of q/k/vt
  for (int i = tid; i < PADM * SNC; i += 256) S_s[i] = 0.f;
  for (int i = tid; i < (PADM - NT) * HD; i += 256) {
    q_s[NT * HD + i] = (bf16)0.f;
    k_s[NT * HD + i] = (bf16)0.f;
  }
  for (int i = tid; i < HD * (PADM - NT); i += 256) {
    int c = i / (PADM - NT), t = NT + i % (PADM - NT);
    vt_s[c * PADM + t] = (bf16)0.f;
  }
  // load q,k (dword copies) and v transposed
  const uint32_t* qsrc = (const uint32_t*)(qg + (size_t)hb * NT * HD);
  const uint32_t* ksrc = (const uint32_t*)(kg + (size_t)hb * NT * HD);
  for (int i = tid; i < NT * HD / 2; i += 256) {
    ((uint32_t*)q_s)[i] = qsrc[i];
    ((uint32_t*)k_s)[i] = ksrc[i];
  }
  const bf16* vsrc = vg + (size_t)hb * NT * HD;
  for (int i = tid; i < NT * HD / 2; i += 256) {
    int t = i >> 5, cp = i & 31;
    bfpair cv;
    cv.u = *(const uint32_t*)(vsrc + t * HD + cp * 2);
    vt_s[(cp * 2) * PADM + t]     = cv.h[0];
    vt_s[(cp * 2 + 1) * PADM + t] = cv.h[1];
  }
  __syncthreads();

  // decomposed rel-pos into zero-padded [208][16] tables:
  //   relh[t][kh] = q[t,:].Rh[h(t),kh,:] ; relw[t][kw] = q[t,:].Rw[w(t),kw,:]
  for (int idx = tid; idx < 2 * PADM * 16; idx += 256) {
    int isw = idx >= PADM * 16;
    int j = isw ? idx - PADM * 16 : idx;
    int t = j >> 4, kk = j & 15;
    float s = 0.f;
    if (t < NT && kk < HH) {
      int coord = isw ? (t % WW) : (t / WW);
      const float* row = (isw ? rel_w : rel_h) + (size_t)(coord - kk + HH - 1) * HD;
      const bf16* qrow = q_s + t * HD;
#pragma unroll 8
      for (int c = 0; c < HD; ++c) s += (float)qrow[c] * row[c];
    }
    (isw ? relw : relh)[j] = s;
  }
  __syncthreads();

  // S = scale * q k^T + rel  (13x13 = 169 tiles round-robin over 8 waves).
  // Padded rows evaluate to exactly 0 (padded cols meet zero V^T in PV) -> unconditional stores.
  const float scale = 0.125f;  // 64^-0.5
  for (int ti = wave; ti < 13 * 13; ti += 8) {
    int mt = ti / 13, nt = ti % 13;
    v8f acc = (v8f){0, 0, 0, 0, 0, 0, 0, 0};
#pragma unroll
    for (int ks = 0; ks < 2; ++ks) {
      const bf16* ar = q_s + (mt * 16 + lmod) * HD + ks * 32 + lhalf * 8;
      v16bf af = make_frag(ar, ar + 16);
      const bf16* br = k_s + (nt * 16 + lmod) * HD + ks * 32 + lhalf * 8;
      acc = wmma_bf16(af, make_frag(br, br + 16), acc);
    }
    int t2 = nt * 16 + lmod;           // lane-constant per tile
    int dv = t2 / 14;                  // 0..14 (pad col 14/15 is zero)
    int md = t2 - dv * 14;
    int rbase = (mt * 16 + 8 * lhalf) * 16;
    float* srow = S_s + (mt * 16 + 8 * lhalf) * SNC + t2;
#pragma unroll
    for (int v = 0; v < 8; ++v)
      srow[v * SNC] = acc[v] * scale + relh[rbase + v * 16 + dv] + relw[rbase + v * 16 + md];
  }
  __syncthreads();

  // softmax stats per row (float4 scans); 1/sum deferred to PV epilogue
  for (int t = tid; t < NT; t += 256) {
    float4* r4 = (float4*)(S_s + t * SNC);  // 49 vec4 = 196 cols
    float m = -1e30f;
    for (int j = 0; j < 49; ++j) {
      float4 f = r4[j];
      m = fmaxf(m, fmaxf(fmaxf(f.x, f.y), fmaxf(f.z, f.w)));
    }
    float s = 0.f;
    for (int j = 0; j < 49; ++j) {
      float4 f = r4[j];
      f.x = __expf(f.x - m); f.y = __expf(f.y - m);
      f.z = __expf(f.z - m); f.w = __expf(f.w - m);
      s += (f.x + f.y) + (f.z + f.w);
      r4[j] = f;
    }
    inv_s[t] = 1.f / s;
  }
  __syncthreads();

  // O = P V : 13 x 4 tiles, K = 224 (7 bf16 WMMA steps), P cvt'd on the fly
  for (int ti = wave; ti < 13 * 4; ti += 8) {
    int mt = ti >> 2, nc = ti & 3;
    v8f acc = (v8f){0, 0, 0, 0, 0, 0, 0, 0};
    for (int ks = 0; ks < 7; ++ks) {
      const float* sr = S_s + (mt * 16 + lmod) * SNC + ks * 32 + lhalf * 8;
      v16bf af;
#pragma unroll
      for (int e = 0; e < 8; ++e) {
        af[e]     = (bf16)sr[e];
        af[8 + e] = (bf16)sr[16 + e];
      }
      const bf16* br = vt_s + (nc * 16 + lmod) * PADM + ks * 32 + lhalf * 8;
      acc = wmma_bf16(af, make_frag(br, br + 16), acc);
    }
#pragma unroll
    for (int v = 0; v < 8; ++v) {
      int t = mt * 16 + v + 8 * lhalf;
      if (t < NT) {
        int c = nc * 16 + lmod;
        out[((size_t)(b * NT + t)) * DIM + head * HD + c] = (bf16)(acc[v] * inv_s[t]);
      }
    }
  }
}

// ---------------------------------------------------------------------------
extern "C" void kernel_launch(void* const* d_in, const int* in_sizes, int n_in,
                              void* d_out, int out_size, void* d_ws, size_t ws_size,
                              hipStream_t stream) {
  const float* x      = (const float*)d_in[0];
  const float* qkv_w  = (const float*)d_in[1];
  const float* qkv_b  = (const float*)d_in[2];
  const float* proj_w = (const float*)d_in[3];
  const float* proj_b = (const float*)d_in[4];
  const float* rel_h  = (const float*)d_in[5];
  const float* rel_w  = (const float*)d_in[6];
  float* out = (float*)d_out;

  const int M = BSZ * NT;  // 12544 rows
  // bf16 workspace carve (~111 MB; attention output reuses the x buffer)
  bf16* ws = (bf16*)d_ws;
  size_t o = 0;
  bf16* xb  = ws + o; o += (size_t)M * DIM;          // packed x / later attn out
  bf16* qwb = ws + o; o += (size_t)DIM * 3 * DIM;    // packed qkv_w
  bf16* pwb = ws + o; o += (size_t)DIM * DIM;        // packed proj_w
  bf16* qb  = ws + o; o += (size_t)NHD * BSZ * NT * HD;
  bf16* kb  = ws + o; o += (size_t)NHD * BSZ * NT * HD;
  bf16* vb  = ws + o; o += (size_t)NHD * BSZ * NT * HD;
  bf16* aob = xb;

  // 1) fp32 -> bf16 packs
  {
    int n2 = M * DIM / 2;
    cvt_f32_bf16<<<(n2 + 255) / 256, 256, 0, stream>>>(x, xb, n2);
    n2 = DIM * 3 * DIM / 2;
    cvt_f32_bf16<<<(n2 + 255) / 256, 256, 0, stream>>>(qkv_w, qwb, n2);
    n2 = DIM * DIM / 2;
    cvt_f32_bf16<<<(n2 + 255) / 256, 256, 0, stream>>>(proj_w, pwb, n2);
  }

  // 2) QKV GEMM with scatter into per-(head,batch) q/k/v
  gemm_bf16<0><<<dim3(M / 128, (3 * DIM) / 128), 256, 0, stream>>>(
      xb, qwb, qkv_b, nullptr, qb, kb, vb, M, 3 * DIM, DIM);

  // 3) fused attention (~294 KB LDS per workgroup, one per head-batch)
  const int SMEM = PADM * HD * 2 * 2      // q_s + k_s
                 + HD * PADM * 2          // vt_s
                 + PADM * SNC * 4         // S_s
                 + 2 * PADM * 16 * 4      // relh + relw (zero-padded)
                 + PADM * 4;              // inv_s  -> 293,696 B
  hipFuncSetAttribute((const void*)attn_fused,
                      hipFuncAttributeMaxDynamicSharedMemorySize, SMEM);
  attn_fused<<<NHD * BSZ, 256, SMEM, stream>>>(qb, kb, vb, rel_h, rel_w, aob);

  // 4) proj GEMM + bias -> fp32 output
  gemm_bf16<1><<<dim3(M / 128, DIM / 128), 256, 0, stream>>>(
      aob, pwb, proj_b, out, nullptr, nullptr, nullptr, M, DIM, DIM);
}
